// ConvNet_30631706755625
// MI455X (gfx1250) — matrix operand order repaired and checked by reading, not compiled, not run on hardware
//
#include <hip/hip_runtime.h>

// ---------------------------------------------------------------------------
// SNN (norse LIF convnet) forward, T=32 steps, B=32.
// conv2 + fc1 run on v_wmma_f32_16x16x32_bf16 with bf16 hi/lo weight split
// (spikes are exactly {0,1} in bf16, so products are exact; f32 accumulate).
// conv2 uses a custom K-permutation so the implicit-im2col A-fragment gather
// compiles to immediate-offset loads from a single per-lane base pointer.
// ---------------------------------------------------------------------------

typedef __attribute__((ext_vector_type(16))) __bf16 v16bf;
typedef __attribute__((ext_vector_type(8)))  __bf16 v8bf;
typedef __attribute__((ext_vector_type(8)))  float  v8f;

#define WMMA_BF16(a, b, c) \
  __builtin_amdgcn_wmma_f32_16x16x32_bf16(false, (a), false, (b), (short)0, (c), false, false)

// LIF constants: DT*TAU_MEM_INV = 0.1, 1 - DT*TAU_SYN_INV = 0.8, V_TH = 1.
__device__ __forceinline__ float lif(float inp, float* v, float* i) {
  float vv = *v, ii = *i;
  float v_dec = vv + 0.1f * ((0.0f - vv) + ii);   // match reference op order
  float z = (v_dec - 1.0f > 0.0f) ? 1.0f : 0.0f;
  *v = (1.0f - z) * v_dec;
  *i = 0.8f * ii + inp;
  return z;
}

// ---------------------------------------------------------------------------
// Weight prep: bf16 hi/lo fragments in the WMMA B-operand lane layout:
//   element j of lane: K = 16*(j/8) + 8*(lane/16) + j%8, N = lane%16
// Fragment stored as [32 lanes][16 bf16] contiguous (1KB per fragment).
// ---------------------------------------------------------------------------
__global__ void prep_fc1_frags(const float* __restrict__ fc1_w,
                               __bf16* __restrict__ bhi, __bf16* __restrict__ blo) {
  int tid = blockIdx.x * 256 + threadIdx.x;
  if (tid >= 169 * 32 * 32) return;          // 169 k-tiles, 32 n-tiles, 32 lanes
  int lane = tid & 31;
  int tmp  = tid >> 5;
  int kt   = tmp % 169;
  int nt   = tmp / 169;
  int l = lane & 15, hf = lane >> 4;
  int n = nt * 16 + l;
  size_t fb = ((size_t)(nt * 169 + kt) * 32 + lane) * 16;
#pragma unroll
  for (int j = 0; j < 16; ++j) {
    int k = kt * 32 + (j >> 3) * 16 + 8 * hf + (j & 7);   // table K order for fc1
    float w = (n < 500) ? fc1_w[(size_t)n * 5408 + k] : 0.0f;
    __bf16 hi = (__bf16)w;
    bhi[fb + j] = hi;
    blo[fb + j] = (__bf16)(w - (float)hi);
  }
}

// conv2 K-permutation: within each 32-wide k-tile, K = chunk*16 + hfb*8 + e.
//   q = (kt*2 + chunk)*8 + e  in [0,208); q<200 -> (i3=q/25, ky=(q%25)/5, kx=q%5)
//   ic = 8*hfb + i3. This makes the A-side gather offset affine in hfb.
__global__ void prep_conv2_frags(const float* __restrict__ conv2_w,
                                 __bf16* __restrict__ bhi, __bf16* __restrict__ blo) {
  int tid = blockIdx.x * 256 + threadIdx.x;
  if (tid >= 13 * 2 * 32) return;            // 13 k-tiles (K=400->416), 2 n-tiles
  int lane = tid & 31;
  int tmp  = tid >> 5;
  int kt   = tmp % 13;
  int nt   = tmp / 13;
  int l = lane & 15;
  int n = nt * 16 + l;                       // < 32
  size_t fb = ((size_t)(nt * 13 + kt) * 32 + lane) * 16;
#pragma unroll
  for (int j = 0; j < 16; ++j) {
    int K = ((j >> 3) << 4) + 8 * (lane >> 4) + (j & 7);      // K within the 32-wide tile
    int chunk = K >> 4;
    int hfb   = (K >> 3) & 1;
    int e     = K & 7;
    int q = (kt * 2 + chunk) * 8 + e;
    float w = 0.0f;
    if (q < 200) {
      int i3 = q / 25;
      int rr = q - i3 * 25;
      int ky = rr / 5;
      int kx = rr - ky * 5;
      int ic = 8 * hfb + i3;
      w = conv2_w[n * 400 + ic * 25 + ky * 5 + kx];   // [32][16*5*5] row-major
    }
    __bf16 hi = (__bf16)w;
    bhi[fb + j] = hi;
    blo[fb + j] = (__bf16)(w - (float)hi);
  }
}

// ---------------------------------------------------------------------------
// Stage A: conv1 (1->16, 5x5, 64->60) + LIF0 + 2x2 maxpool -> s1 [B,16,30,30] bf16
// ---------------------------------------------------------------------------
__global__ void conv1_lif_pool(const float* __restrict__ x,
                               const float* __restrict__ w1,
                               const float* __restrict__ b1,
                               float* __restrict__ v0, float* __restrict__ i0,
                               __bf16* __restrict__ s1, int t) {
  int bo = blockIdx.x;                 // b*16 + oc
  int b  = bo >> 4;
  int oc = bo & 15;
  float w[25];
#pragma unroll
  for (int q = 0; q < 25; ++q) w[q] = w1[oc * 25 + q];
  float bias = b1[oc];
  const float* xt = x + ((size_t)t * 32 + b) * 4096;   // 64*64, C=1

  for (int p = threadIdx.x; p < 900; p += 256) {       // 30*30 pooled outputs
    int py = p / 30, px = p % 30;
    float xv[6][6];
#pragma unroll
    for (int iy = 0; iy < 6; ++iy)
#pragma unroll
      for (int ix = 0; ix < 6; ++ix)
        xv[iy][ix] = xt[(2 * py + iy) * 64 + (2 * px + ix)];

    float acc[2][2] = {{bias, bias}, {bias, bias}};
#pragma unroll
    for (int ky = 0; ky < 5; ++ky)
#pragma unroll
      for (int kx = 0; kx < 5; ++kx) {
        float ww = w[ky * 5 + kx];
        acc[0][0] = fmaf(ww, xv[ky][kx],     acc[0][0]);
        acc[0][1] = fmaf(ww, xv[ky][kx + 1], acc[0][1]);
        acc[1][0] = fmaf(ww, xv[ky + 1][kx], acc[1][0]);
        acc[1][1] = fmaf(ww, xv[ky + 1][kx + 1], acc[1][1]);
      }

    float zmax = 0.0f;
#pragma unroll
    for (int dy = 0; dy < 2; ++dy)
#pragma unroll
      for (int dx = 0; dx < 2; ++dx) {
        int si = (bo * 60 + (2 * py + dy)) * 60 + (2 * px + dx);
        float z = lif(acc[dy][dx], &v0[si], &i0[si]);
        zmax = fmaxf(zmax, z);
      }
    s1[bo * 900 + p] = (__bf16)zmax;
  }
}

// ---------------------------------------------------------------------------
// Stage B1: conv2 as implicit-im2col WMMA GEMM with the permuted K order.
//   A row m = (b, y, x) over 26x26; per-lane gather base:
//     ab = s1 + (b*16 + 8*hf)*900 + y*30 + x
//   element (chunk,e): offset i3*900 + ky*30 + kx -- all compile-time constants.
//   D = 10*(A@W + bias) -> a2 [B,32,26,26] fp32.
// ---------------------------------------------------------------------------
__global__ void conv2_wmma(const __bf16* __restrict__ s1,
                           const __bf16* __restrict__ bch,
                           const __bf16* __restrict__ bcl,
                           const float* __restrict__ c2b,
                           float* __restrict__ a2) {
  int mt   = blockIdx.x;
  int lane = threadIdx.x;
  int l = lane & 15, hf = lane >> 4;

  int m = mt * 16 + l;                 // A-fragment row for this lane (M = lane&15)
  int b = m / 676;
  int p = m - b * 676;
  int y = p / 26;
  int xx = p - y * 26;

  const __bf16* ab = s1 + ((size_t)b * 16 + 8 * hf) * 900 + y * 30 + xx;
  const __bf16* bh = bch + (size_t)lane * 16;
  const __bf16* blw = bcl + (size_t)lane * 16;

  v8f acc0 = {0.f,0.f,0.f,0.f,0.f,0.f,0.f,0.f};
  v8f acc1 = {0.f,0.f,0.f,0.f,0.f,0.f,0.f,0.f};

#pragma unroll
  for (int kt = 0; kt < 13; ++kt) {
    v16bf a;
#pragma unroll
    for (int ch = 0; ch < 2; ++ch)
#pragma unroll
      for (int e = 0; e < 8; ++e) {
        int q = (kt * 2 + ch) * 8 + e;           // compile-time after unroll
        float v = 0.0f;
        if (q < 200) {
          int i3 = q / 25;
          int rr = q - i3 * 25;
          int ky = rr / 5;
          int kx = rr - ky * 5;
          v = (float)ab[i3 * 900 + ky * 30 + kx]; // immediate-offset load
        }
        a[ch * 8 + e] = (__bf16)v;
      }
    v16bf b0h = *(const v16bf*)(bh  + (size_t)kt * 512);
    v16bf b0l = *(const v16bf*)(blw + (size_t)kt * 512);
    v16bf b1h = *(const v16bf*)(bh  + (size_t)(13 + kt) * 512);
    v16bf b1l = *(const v16bf*)(blw + (size_t)(13 + kt) * 512);
    acc0 = WMMA_BF16(a, b0h, acc0);
    acc0 = WMMA_BF16(a, b0l, acc0);
    acc1 = WMMA_BF16(a, b1h, acc1);
    acc1 = WMMA_BF16(a, b1l, acc1);
  }

  // Epilogue: D layout is N=lane&15, M=r+8*hf. Apply 10*(acc+bias), scatter.
#pragma unroll
  for (int nt = 0; nt < 2; ++nt) {
    const v8f& acc = nt ? acc1 : acc0;
    int n = nt * 16 + l;
    float bias = c2b[n];
#pragma unroll
    for (int r = 0; r < 8; ++r) {
      int mm = mt * 16 + r + 8 * hf;
      int bb = mm / 676;
      int pp = mm - bb * 676;
      int yy = pp / 26;
      int xo = pp - yy * 26;
      a2[((bb * 32 + n) * 26 + yy) * 26 + xo] = 10.0f * (acc[r] + bias);
    }
  }
}

// ---------------------------------------------------------------------------
// Stage B2: LIF1 on a2 + 2x2 maxpool -> s2 [B,5408] bf16.
// ---------------------------------------------------------------------------
__global__ void lif1_pool(const float* __restrict__ a2,
                          float* __restrict__ v1, float* __restrict__ i1,
                          __bf16* __restrict__ s2) {
  int idx = blockIdx.x * 256 + threadIdx.x;
  if (idx >= 32 * 5408) return;
  int b = idx / 5408;
  int f = idx - b * 5408;
  int oc = f / 169;
  int r  = f - oc * 169;
  int ph = r / 13;
  int pw = r - ph * 13;
  float zmax = 0.0f;
#pragma unroll
  for (int dy = 0; dy < 2; ++dy)
#pragma unroll
    for (int dx = 0; dx < 2; ++dx) {
      int si = ((b * 32 + oc) * 26 + (2 * ph + dy)) * 26 + (2 * pw + dx);
      float z = lif(a2[si], &v1[si], &i1[si]);
      zmax = fmaxf(zmax, z);
    }
  s2[idx] = (__bf16)zmax;
}

// ---------------------------------------------------------------------------
// Stage C1: fc1 GEMM [32 x 5408] @ [5408 x 512] via WMMA + LIF2 (+relu==z).
// ---------------------------------------------------------------------------
__global__ void fc1_wmma_lif(const __bf16* __restrict__ s2,
                             const __bf16* __restrict__ bfh,
                             const __bf16* __restrict__ bfl,
                             const float* __restrict__ fc1b,
                             float* __restrict__ v2, float* __restrict__ i2,
                             float* __restrict__ zr) {
  int tile = blockIdx.x;
  int mt = tile & 1;
  int nt = tile >> 1;
  int lane = threadIdx.x;
  int l = lane & 15, hf = lane >> 4;

  const __bf16* s2row = s2 + (size_t)(mt * 16 + l) * 5408;
  v8f acc = {0.f,0.f,0.f,0.f,0.f,0.f,0.f,0.f};

#pragma unroll 2
  for (int kt = 0; kt < 169; ++kt) {
    v8bf c0 = *(const v8bf*)(s2row + kt * 32 + 8 * hf);
    v8bf c1 = *(const v8bf*)(s2row + kt * 32 + 16 + 8 * hf);
    v16bf a;
#pragma unroll
    for (int j = 0; j < 8; ++j) { a[j] = c0[j]; a[j + 8] = c1[j]; }

    size_t fb = ((size_t)(nt * 169 + kt) * 32 + lane) * 16;
    if (kt + 1 < 169)
      __builtin_prefetch(bfh + fb + 512, 0, 1);   // next k-tile fragment
    v16bf bh = *(const v16bf*)(bfh + fb);
    v16bf bl = *(const v16bf*)(bfl + fb);
    acc = WMMA_BF16(a, bh, acc);
    acc = WMMA_BF16(a, bl, acc);
  }

  int n = nt * 16 + l;                 // D: N = lane&15
  if (n < 500) {
    float bias = fc1b[n];
#pragma unroll
    for (int r = 0; r < 8; ++r) {
      int bb = mt * 16 + r + 8 * hf;   // batch row
      int si = bb * 500 + n;
      float z = lif(acc[r] + bias, &v2[si], &i2[si]);
      zr[si] = z;                      // relu(z) == z for spikes
    }
  }
}

// ---------------------------------------------------------------------------
// Stage C2: LI readout. v_out = vo + 0.1*(-vo + io); io = 0.8*io + zr@W^T.
// ---------------------------------------------------------------------------
__global__ void li_readout(const float* __restrict__ zr,
                           const float* __restrict__ outw,
                           float* __restrict__ vo, float* __restrict__ io,
                           float* __restrict__ out, int t) {
  int tid = threadIdx.x;
  if (tid >= 320) return;
  int b = tid / 10;
  int c = tid - b * 10;
  float vov = vo[tid], iov = io[tid];
  float vnew = vov + 0.1f * ((0.0f - vov) + iov);
  float dot = 0.0f;
  const float* zb = zr + b * 500;
  const float* wc = outw + c * 500;
  for (int f = 0; f < 500; ++f) dot = fmaf(zb[f], wc[f], dot);
  vo[tid] = vnew;
  io[tid] = 0.8f * iov + dot;
  out[(size_t)t * 320 + tid] = vnew;
}

// ---------------------------------------------------------------------------
extern "C" void kernel_launch(void* const* d_in, const int* in_sizes, int n_in,
                              void* d_out, int out_size, void* d_ws, size_t ws_size,
                              hipStream_t stream) {
  const float* x    = (const float*)d_in[0];   // [32,32,1,64,64]
  const float* c1w  = (const float*)d_in[1];   // [16,1,5,5]
  const float* c1b  = (const float*)d_in[2];   // [16]
  const float* c2w  = (const float*)d_in[3];   // [32,16,5,5]
  const float* c2b  = (const float*)d_in[4];   // [32]
  const float* f1w  = (const float*)d_in[5];   // [500,5408]
  const float* f1b  = (const float*)d_in[6];   // [500]
  const float* ow   = (const float*)d_in[7];   // [10,500]
  float* out = (float*)d_out;                  // [32,32,10]

  char* base = (char*)d_ws;
  size_t off = 0;
  auto take = [&](size_t bytes) -> size_t {
    size_t r = off;
    off = (off + bytes + 255) & ~(size_t)255;
    return r;
  };
  // persistent LIF/LI state (zeroed each launch)
  size_t o_v0 = take((size_t)1843200 * 4);
  size_t o_i0 = take((size_t)1843200 * 4);
  size_t o_v1 = take((size_t)692224 * 4);
  size_t o_i1 = take((size_t)692224 * 4);
  size_t o_v2 = take((size_t)16000 * 4);
  size_t o_i2 = take((size_t)16000 * 4);
  size_t o_vo = take((size_t)320 * 4);
  size_t o_io = take((size_t)320 * 4);
  size_t state_bytes = off;
  // per-step transients
  size_t o_s1  = take((size_t)460800 * 2);           // bf16 pooled spikes stage1
  size_t o_s2  = take((size_t)32 * 5408 * 2);        // bf16 pooled spikes stage2
  size_t o_a2  = take((size_t)692224 * 4);           // conv2 pre-activation
  size_t o_zr  = take((size_t)16000 * 4);            // fc1 spikes
  // bf16 hi/lo weight fragments
  size_t o_bfh = take((size_t)169 * 32 * 512 * 2);
  size_t o_bfl = take((size_t)169 * 32 * 512 * 2);
  size_t o_bch = take((size_t)13 * 2 * 512 * 2);
  size_t o_bcl = take((size_t)13 * 2 * 512 * 2);
  (void)ws_size; (void)in_sizes; (void)n_in; (void)out_size;

  float*  v0  = (float*)(base + o_v0);
  float*  i0  = (float*)(base + o_i0);
  float*  v1  = (float*)(base + o_v1);
  float*  i1  = (float*)(base + o_i1);
  float*  v2  = (float*)(base + o_v2);
  float*  i2  = (float*)(base + o_i2);
  float*  vo  = (float*)(base + o_vo);
  float*  io  = (float*)(base + o_io);
  __bf16* s1  = (__bf16*)(base + o_s1);
  __bf16* s2  = (__bf16*)(base + o_s2);
  float*  a2  = (float*)(base + o_a2);
  float*  zr  = (float*)(base + o_zr);
  __bf16* bfh = (__bf16*)(base + o_bfh);
  __bf16* bfl = (__bf16*)(base + o_bfl);
  __bf16* bch = (__bf16*)(base + o_bch);
  __bf16* bcl = (__bf16*)(base + o_bcl);

  hipMemsetAsync(d_ws, 0, state_bytes, stream);

  prep_fc1_frags<<<(169 * 32 * 32 + 255) / 256, 256, 0, stream>>>(f1w, bfh, bfl);
  prep_conv2_frags<<<(13 * 2 * 32 + 255) / 256, 256, 0, stream>>>(c2w, bch, bcl);

  for (int t = 0; t < 32; ++t) {
    conv1_lif_pool<<<32 * 16, 256, 0, stream>>>(x, c1w, c1b, v0, i0, s1, t);
    conv2_wmma<<<1352, 32, 0, stream>>>(s1, bch, bcl, c2b, a2);
    lif1_pool<<<(32 * 5408 + 255) / 256, 256, 0, stream>>>(a2, v1, i1, s2);
    fc1_wmma_lif<<<64, 32, 0, stream>>>(s2, bfh, bfl, f1b, v2, i2, zr);
    li_readout<<<1, 320, 0, stream>>>(zr, ow, vo, io, out, t);
  }
}
